// Attention_68410239091507
// MI455X (gfx1250) — matrix operand order repaired and checked by reading, not compiled, not run on hardware
//
#include <hip/hip_runtime.h>
#include <hip/hip_bf16.h>

typedef __attribute__((ext_vector_type(16))) __bf16 v16bf;
typedef __attribute__((ext_vector_type(8)))  __bf16 v8bf;
typedef __attribute__((ext_vector_type(8)))  float  v8f;

#define BB   4
#define SS   4096
#define DD   256
#define LL   8921
#define LPAD 8928         // round up to 16
#define NLT  558          // LPAD/16 L-tiles
#define NW   8
#define NTHREADS 256

__device__ __forceinline__ unsigned short f2bf(float f) {
    union { float f; unsigned u; } v; v.f = f;
    unsigned r = v.u + 0x7FFFu + ((v.u >> 16) & 1u);   // round-to-nearest-even
    return (unsigned short)(r >> 16);
}

// Convert x -> bf16 row-major [b][s][d] AND bf16 transposed [b][d][s]
__global__ void cvt_x_kernel(const float* __restrict__ x,
                             unsigned short* __restrict__ xbf,
                             unsigned short* __restrict__ xT) {
    int i = blockIdx.x * blockDim.x + threadIdx.x;
    if (i >= BB * SS * DD) return;
    unsigned short h = f2bf(x[i]);
    xbf[i] = h;
    int b   = i / (SS * DD);
    int rem = i % (SS * DD);
    int s   = rem / DD;
    int d   = rem % DD;
    xT[((size_t)b * DD + d) * SS + s] = h;
}

// Convert U -> bf16, zero-padded to LPAD rows
__global__ void cvt_u_kernel(const float* __restrict__ U,
                             unsigned short* __restrict__ ubf) {
    int i = blockIdx.x * blockDim.x + threadIdx.x;
    if (i >= LPAD * DD) return;
    int l = i / DD;
    ubf[i] = (l < LL) ? f2bf(U[i]) : (unsigned short)0;
}

// Fused: scores(16xS in LDS) -> softmax(S) -> alpha out + pooled out
extern "C" __global__ __launch_bounds__(NTHREADS)
void attn_fused(const unsigned short* __restrict__ xbf,  // [B][S][D] bf16
                const unsigned short* __restrict__ xT,   // [B][D][S] bf16
                const unsigned short* __restrict__ ubf,  // [LPAD][D] bf16
                float* __restrict__ out,                 // [B][L][D]
                float* __restrict__ alpha)               // [B][L][S]
{
    extern __shared__ float lds[];                       // 16 * SS floats (256 KB)
    const int wg    = blockIdx.x;
    const int b     = wg / NLT;
    const int l0    = (wg % NLT) * 16;
    const int tid   = threadIdx.x;
    const int wave  = tid >> 5;
    const int lane  = tid & 31;
    const int lhalf = lane >> 4;                         // K half-select
    const int lmod  = lane & 15;                         // row (A) / col (B,D)

    // ---------------- GEMM1: scores[16][SS] = U_tile(16xD) * x[b]^T ----------------
    // Hoist all 8 A fragments (K = 0..255 in steps of 32) for this L-tile.
    v16bf afrag[8];
    {
        const unsigned short* urow = ubf + (size_t)(l0 + lmod) * DD;
#pragma unroll
        for (int kk = 0; kk < 8; ++kk) {
            union { v16bf v; v8bf h[2]; } ua;
            ua.h[0] = *(const v8bf*)(urow + kk * 32 + lhalf * 16);
            ua.h[1] = *(const v8bf*)(urow + kk * 32 + lhalf * 16 + 8);
            afrag[kk] = ua.v;
        }
    }
    const unsigned short* xb = xbf + (size_t)b * SS * DD;
    for (int t = 0; t < 32; ++t) {                       // 8 waves x 32 tiles = S/16
        const int s0 = (wave * 32 + t) * 16;
        v8f acc = {};
#pragma unroll
        for (int kk = 0; kk < 8; ++kk) {
            // B[k][n] = x[b][s0+n][k]: column n is a contiguous x row -> 32B load
            v16bf bfrag = *(const v16bf*)(xb + (size_t)(s0 + lmod) * DD + kk * 32 + lhalf * 16);
            acc = __builtin_amdgcn_wmma_f32_16x16x32_bf16(
                      false, afrag[kk], false, bfrag, (short)0, acc, false, false);
        }
#pragma unroll
        for (int r = 0; r < 8; ++r)                      // D-layout: row = r + 8*lhalf
            lds[(size_t)(r + 8 * lhalf) * SS + s0 + lmod] = acc[r];
    }
    __syncthreads();

    // -------- softmax over S per row; write alpha (f32) + pack bf16 in-place in LDS ----
#pragma unroll
    for (int rr = 0; rr < 2; ++rr) {                     // 8 waves x 2 rows = 16
        const int m = wave * 2 + rr;
        float* row = lds + (size_t)m * SS;

        float mx = -3.4e38f;
        for (int i = lane * 4; i < SS; i += 128) {
            float4 v = *(const float4*)(row + i);
            mx = fmaxf(mx, fmaxf(fmaxf(v.x, v.y), fmaxf(v.z, v.w)));
        }
#pragma unroll
        for (int o = 16; o >= 1; o >>= 1) mx = fmaxf(mx, __shfl_xor(mx, o, 32));

        float sum = 0.f;
        for (int i = lane * 4; i < SS; i += 128) {
            float4 v = *(float4*)(row + i);
            v.x = __expf(v.x - mx); v.y = __expf(v.y - mx);
            v.z = __expf(v.z - mx); v.w = __expf(v.w - mx);
            *(float4*)(row + i) = v;
            sum += (v.x + v.y) + (v.z + v.w);
        }
#pragma unroll
        for (int o = 16; o >= 1; o >>= 1) sum += __shfl_xor(sum, o, 32);
        const float inv = 1.0f / sum;

        unsigned short* rowbf = (unsigned short*)row;    // bf16 alpha, packed in place
        const int l = l0 + m;
        float* arow = (l < LL) ? (alpha + ((size_t)b * LL + l) * SS) : (float*)0;
        for (int i = lane * 4; i < SS; i += 128) {
            float4 v = *(const float4*)(row + i);        // ds_load before ds_store: safe
            v.x *= inv; v.y *= inv; v.z *= inv; v.w *= inv;
            if (arow) *(float4*)(arow + i) = v;          // b128 store on dominant stream
            union { unsigned short us[4]; uint2 u2; } p;
            p.us[0] = f2bf(v.x); p.us[1] = f2bf(v.y);
            p.us[2] = f2bf(v.z); p.us[3] = f2bf(v.w);
            *(uint2*)(rowbf + i) = p.u2;                 // 8B aligned (i % 4 == 0)
        }
    }
    __syncthreads();

    // ---------------- GEMM2: out[16][DD] = alpha_tile(16xS) * x[b](SxD) ----------------
    v8f acc2[2] = {};                                    // d-tiles: d0 = wave*32 + t*16
    const unsigned short* arowbf = (const unsigned short*)(lds + (size_t)lmod * SS);
    const unsigned short* xtb = xT + (size_t)b * DD * SS;
    const unsigned short* brow0 = xtb + (size_t)(wave * 32 + lmod) * SS;
    const unsigned short* brow1 = xtb + (size_t)(wave * 32 + 16 + lmod) * SS;
    for (int s0 = 0; s0 < SS; s0 += 32) {
        // A fragment straight from packed-bf16 LDS: two ds_load_b128, zero VALU
        union { v16bf v; v8bf h[2]; } ua;
        const int kb = s0 + lhalf * 16;
        ua.h[0] = *(const v8bf*)(arowbf + kb);
        ua.h[1] = *(const v8bf*)(arowbf + kb + 8);
        v16bf b0 = *(const v16bf*)(brow0 + s0 + lhalf * 16);
        acc2[0] = __builtin_amdgcn_wmma_f32_16x16x32_bf16(
                      false, ua.v, false, b0, (short)0, acc2[0], false, false);
        v16bf b1 = *(const v16bf*)(brow1 + s0 + lhalf * 16);
        acc2[1] = __builtin_amdgcn_wmma_f32_16x16x32_bf16(
                      false, ua.v, false, b1, (short)0, acc2[1], false, false);
    }
#pragma unroll
    for (int t = 0; t < 2; ++t) {
        const int d0 = wave * 32 + t * 16;
#pragma unroll
        for (int r = 0; r < 8; ++r) {
            const int l = l0 + r + 8 * lhalf;
            if (l < LL)
                out[((size_t)b * LL + l) * DD + d0 + lmod] = acc2[t][r];
        }
    }
}

extern "C" void kernel_launch(void* const* d_in, const int* in_sizes, int n_in,
                              void* d_out, int out_size, void* d_ws, size_t ws_size,
                              hipStream_t stream) {
    (void)in_sizes; (void)n_in; (void)out_size; (void)ws_size;
    const float* x = (const float*)d_in[0];   // [B][S][D]
    const float* U = (const float*)d_in[1];   // [L][D]
    float* out   = (float*)d_out;                         // [B][L][D] first
    float* alpha = out + (size_t)BB * LL * DD;            // then [B][L][S]

    char* ws = (char*)d_ws;
    unsigned short* xbf = (unsigned short*)ws;                                  // 8 MiB
    unsigned short* xT  = (unsigned short*)(ws + (size_t)BB * SS * DD * 2);     // 8 MiB
    unsigned short* ubf = (unsigned short*)(ws + (size_t)BB * SS * DD * 4);     // 4.4 MiB

    {
        int n = BB * SS * DD;
        cvt_x_kernel<<<(n + 255) / 256, 256, 0, stream>>>(x, xbf, xT);
    }
    {
        int n = LPAD * DD;
        cvt_u_kernel<<<(n + 255) / 256, 256, 0, stream>>>(U, ubf);
    }
    const size_t lds_bytes = (size_t)16 * SS * sizeof(float);   // 256 KB (of 320 KB/WGP)
    hipFuncSetAttribute((const void*)attn_fused,
                        hipFuncAttributeMaxDynamicSharedMemorySize, (int)lds_bytes);
    attn_fused<<<BB * NLT, NTHREADS, lds_bytes, stream>>>(xbf, xT, ubf, out, alpha);
}
